// AdaptiveVectorQuantizer_1898375545426
// MI455X (gfx1250) — matrix operand order repaired and checked by reading, not compiled, not run on hardware
//
#include <hip/hip_runtime.h>
#include <hip/hip_bf16.h>

#define NUM_EMB   512
#define EMB_DIM   128
#define N_ROWS    (64 * 2048)          // 131072
#define Q_ELEMS   (N_ROWS * EMB_DIM)   // 16777216
#define LDS_ROW_BYTES 272              // 256B row + 16B pad -> bank-conflict free
#define LDS_BYTES (NUM_EMB * LDS_ROW_BYTES)  // 139264 <= 320KB WGP LDS

typedef __attribute__((ext_vector_type(16))) __bf16 v16bf;
typedef __attribute__((ext_vector_type(8)))  float  v8f;

struct Bytes32 { uint4 lo, hi; };

// ---------------------------------------------------------------------------
// Prep: f32 codebook -> bf16 codebook in ws; ew[k] = { ||e_k||^2 , hr_weight }
// ---------------------------------------------------------------------------
__global__ void vq_prep(const float* __restrict__ emb,
                        const float* __restrict__ scaling,
                        unsigned short* __restrict__ cb_bf16,
                        float2* __restrict__ ew) {
    int k = blockIdx.x * blockDim.x + threadIdx.x;
    if (k >= NUM_EMB) return;
    float sum = 0.f;
    const float* row = emb + (size_t)k * EMB_DIM;
    unsigned short* dst = cb_bf16 + (size_t)k * EMB_DIM;
    #pragma unroll 8
    for (int d = 0; d < EMB_DIM; ++d) {
        float f = row[d];
        sum += f * f;
        __bf16 b = (__bf16)f;
        dst[d] = __builtin_bit_cast(unsigned short, b);
    }
    float hr = 40.0f + 140.0f * (float)k / 511.0f;
    float w  = 1.0f + scaling[k] * ((hr - 100.0f) / 70.0f);
    ew[k] = make_float2(sum, w);
}

// ---------------------------------------------------------------------------
// Main: per wave: 16 rows x all 512 codes via v_wmma_f32_16x16x32_bf16,
// double-buffered LDS B-fragments to hide ds latency behind WMMA.
// ---------------------------------------------------------------------------
__global__ void vq_main(const float* __restrict__ x,
                        const float* __restrict__ emb_f32,
                        const unsigned short* __restrict__ cb_bf16,
                        const float2* __restrict__ ew,
                        float* __restrict__ outq,
                        float* __restrict__ outi,
                        float* __restrict__ loss) {
    extern __shared__ char lds[];

    // ---- stage bf16 codebook into LDS with padded rows (128KB -> 136KB) ----
    {
        const uint4* src = (const uint4*)cb_bf16;   // 512 rows * 16 uint4
        for (int i = threadIdx.x; i < NUM_EMB * 16; i += 256) {
            int code = i >> 4, part = i & 15;
            *(uint4*)(lds + code * LDS_ROW_BYTES + part * 16) = src[i];
        }
    }
    __syncthreads();

    const int lane = threadIdx.x & 31;
    const int wv   = threadIdx.x >> 5;       // wave id 0..7
    const int half = lane >> 4;              // 0 | 1
    const int l15  = lane & 15;
    const int m0   = blockIdx.x * 128 + wv * 16;   // first of 16 rows for this wave

    // ---- build A fragments (16x32 bf16 per chunk) + f32 ||x||^2 ----
    // lane layout: M = lane&15 ; K = {b..b+15}, b = half*16 (+32*chunk)
    const float* xrow = x + (size_t)(m0 + l15) * EMB_DIM;
    const int bK = half * 16;
    v16bf afrag[4];
    float x2p = 0.f;
    #pragma unroll
    for (int c = 0; c < 4; ++c) {
        float f0[8], f1[8];
        const float4 a0 = *(const float4*)(xrow + c * 32 + bK);
        const float4 a1 = *(const float4*)(xrow + c * 32 + bK + 4);
        const float4 a2 = *(const float4*)(xrow + c * 32 + bK + 8);
        const float4 a3 = *(const float4*)(xrow + c * 32 + bK + 12);
        f0[0]=a0.x; f0[1]=a0.y; f0[2]=a0.z; f0[3]=a0.w;
        f0[4]=a1.x; f0[5]=a1.y; f0[6]=a1.z; f0[7]=a1.w;
        f1[0]=a2.x; f1[1]=a2.y; f1[2]=a2.z; f1[3]=a2.w;
        f1[4]=a3.x; f1[5]=a3.y; f1[6]=a3.z; f1[7]=a3.w;
        #pragma unroll
        for (int i = 0; i < 8; ++i) {
            x2p += f0[i] * f0[i] + f1[i] * f1[i];
            afrag[c][i]     = (__bf16)f0[i];
            afrag[c][8 + i] = (__bf16)f1[i];
        }
    }
    // row M's norm is split across lanes M and M+16
    x2p += __shfl_xor(x2p, 16, 32);
    float x2m[8];
    #pragma unroll
    for (int j = 0; j < 8; ++j)
        x2m[j] = __shfl(x2p, j + half * 8, 32);   // C-slot (lane,vgpr j): M = j + 8*half

    // ---- running weighted argmin state ----
    float bestS[8];
    int   bestI[8];
    #pragma unroll
    for (int j = 0; j < 8; ++j) { bestS[j] = 3.4e38f; bestI[j] = 0; }

    // B-fragment loader: 8 ds_load_b128 into distinct regs (clause-able)
    auto loadB = [&](v16bf bf[4], int code) {
        #pragma unroll
        for (int c = 0; c < 4; ++c) {
            const size_t off = (size_t)code * LDS_ROW_BYTES + (size_t)(half * 16 + c * 32) * 2;
            Bytes32 raw;
            raw.lo = *(const uint4*)(lds + off);
            raw.hi = *(const uint4*)(lds + off + 16);
            bf[c] = __builtin_bit_cast(v16bf, raw);
        }
    };

    auto tileCompute = [&](const v16bf bf[4], float2 ewv, int code) {
        v8f acc = {0.f,0.f,0.f,0.f,0.f,0.f,0.f,0.f};
        #pragma unroll
        for (int c = 0; c < 4; ++c)
            acc = __builtin_amdgcn_wmma_f32_16x16x32_bf16(
                false, afrag[c], false, bf[c], (short)0, acc, false, false);
        #pragma unroll
        for (int j = 0; j < 8; ++j) {
            float dist  = x2m[j] + ewv.x - 2.0f * acc[j];
            float score = dist * ewv.y;
            if (score < bestS[j]) { bestS[j] = score; bestI[j] = code; }
        }
    };

    // ---- double-buffered tile loop over 32 code tiles ----
    v16bf bufA[4], bufB[4];
    float2 ewA, ewB;
    {
        int c0 = l15;                         // tile 0
        loadB(bufA, c0);
        ewA = ew[c0];
    }
    for (int t = 0; t < 32; t += 2) {
        const int codeA = t * 16 + l15;
        const int codeB = min((t + 1) * 16 + l15, NUM_EMB - 1);
        loadB(bufB, codeB);                   // prefetch tile t+1
        ewB = ew[codeB];
        tileCompute(bufA, ewA, codeA);        // compute tile t

        const int codeA2 = min((t + 2) * 16 + l15, NUM_EMB - 1);
        loadB(bufA, codeA2);                  // prefetch tile t+2 (clamped ghost)
        ewA = ew[codeA2];
        tileCompute(bufB, ewB, (t + 1) * 16 + l15);  // compute tile t+1
    }

    // ---- butterfly min-reduce across the 16 lanes of each half ----
    #pragma unroll
    for (int s = 1; s < 16; s <<= 1) {
        #pragma unroll
        for (int j = 0; j < 8; ++j) {
            float oS = __shfl_xor(bestS[j], s, 32);
            int   oI = __shfl_xor(bestI[j], s, 32);
            if (oS < bestS[j] || (oS == bestS[j] && oI < bestI[j])) {
                bestS[j] = oS; bestI[j] = oI;
            }
        }
    }

    // ---- indices + loss (lanes 0 and 16 hold rows 0..7 / 8..15) ----
    if (l15 == 0) {
        float dsum = 0.f;
        #pragma unroll
        for (int j = 0; j < 8; ++j) {
            int r = m0 + j + half * 8;
            outi[r] = (float)bestI[j];
            dsum += bestS[j] / ew[bestI[j]].y;   // recover unweighted distance
        }
        // loss = (1 + 0.6) * sum(dist) / (N*D)
        atomicAdd(loss, dsum * (1.6f / (float)Q_ELEMS));
    }

    // ---- quantized output: gather winning code rows (1 float4/lane/row) ----
    #pragma unroll
    for (int M = 0; M < 16; ++M) {
        int idxM = __shfl(bestI[M & 7], (M >> 3) * 16, 32);
        float4 v = *((const float4*)(emb_f32 + (size_t)idxM * EMB_DIM) + lane);
        *((float4*)(outq + (size_t)(m0 + M) * EMB_DIM) + lane) = v;
    }
}

// ---------------------------------------------------------------------------
extern "C" void kernel_launch(void* const* d_in, const int* in_sizes, int n_in,
                              void* d_out, int out_size, void* d_ws, size_t ws_size,
                              hipStream_t stream) {
    const float* x    = (const float*)d_in[0];   // [64,2048,128] f32
    const float* emb  = (const float*)d_in[1];   // [512,128] f32
    const float* scal = (const float*)d_in[2];   // [512] f32

    float* outq = (float*)d_out;                 // quantized_st: 16777216 floats
    float* loss = outq + Q_ELEMS;                // 1 float
    float* outi = loss + 1;                      // 131072 floats (indices)

    unsigned short* cb = (unsigned short*)d_ws;               // 512*128 bf16
    float2* ewp = (float2*)((char*)d_ws + (size_t)NUM_EMB * EMB_DIM * 2);

    hipMemsetAsync(loss, 0, sizeof(float), stream);
    vq_prep<<<2, 256, 0, stream>>>(emb, scal, cb, ewp);
    vq_main<<<N_ROWS / 128, 256, LDS_BYTES, stream>>>(
        x, emb, cb, ewp, outq, outi, loss);
}
